// AttnDecoderRNN_54271206752589
// MI455X (gfx1250) — matrix operand order repaired and checked by reading, not compiled, not run on hardware
//
#include <hip/hip_runtime.h>
#include <hip/hip_bf16.h>
#include <math.h>

// ---------- problem constants ----------
#define Bn 32
#define Sn 256
#define Tn 64
#define Hn 512
#define Vn 32000
#define SOS 1

// ---------- WMMA types (CDNA5 gfx1250, wave32) ----------
typedef __bf16 bf16v16 __attribute__((ext_vector_type(16)));
typedef float  f32v8   __attribute__((ext_vector_type(8)));

union FragBF { bf16v16 v; uint4 q[2]; unsigned short u[16]; };
union AccF   { f32v8 v; float f[8]; };

__device__ __forceinline__ unsigned short f2bf(float f) {
    unsigned int u = __float_as_uint(f);
    return (unsigned short)((u + 0x7FFFu + ((u >> 16) & 1u)) >> 16);
}

// ---------- fp32 -> bf16 convert ----------
__global__ void k_f32_to_bf16(const float* __restrict__ in,
                              unsigned short* __restrict__ out, int n) {
    int i = blockIdx.x * blockDim.x + threadIdx.x;
    if (i < n) out[i] = f2bf(in[i]);
}

__global__ void k_copy_f32(const float* __restrict__ in, float* __restrict__ out, int n) {
    int i = blockIdx.x * blockDim.x + threadIdx.x;
    if (i < n) out[i] = in[i];
}

// ---------- WMMA bf16 GEMM:  C[M,N] = A[M,K] @ W[N,K]^T + bias[N]  ----------
// A row-major bf16, W row-major (N,K) bf16 (PyTorch-style weight), C fp32.
// Wave tile: 16(M) x 64(N), K stepped by 32 via v_wmma_f32_16x16x32_bf16.
// M,K multiples of 16/32; N multiple of 64.
__global__ void k_wmma_gemm_bf16(const unsigned short* __restrict__ A,
                                 const unsigned short* __restrict__ W,
                                 const float* __restrict__ bias,
                                 float* __restrict__ C,
                                 int M, int N, int K) {
    const int NT = 4; // 4 x 16 columns per wave
    int wave = (blockIdx.x * blockDim.x + threadIdx.x) >> 5;
    int lane = threadIdx.x & 31;
    int ntiles = N / (16 * NT);
    int mtiles = M / 16;
    if (wave >= mtiles * ntiles) return;
    int m0 = (wave / ntiles) * 16;
    int n0 = (wave % ntiles) * (16 * NT);

    int half = lane >> 4;   // K half-selector
    int mr   = lane & 15;   // A row within tile / B col within tile / C col

    AccF acc[NT];
#pragma unroll
    for (int j = 0; j < NT; ++j)
#pragma unroll
        for (int r = 0; r < 8; ++r) acc[j].f[r] = 0.0f;

    const unsigned short* arow = A + (size_t)(m0 + mr) * K;
    for (int k0 = 0; k0 < K; k0 += 32) {
        // A fragment: lanes 0-15 hold K {k0..k0+7, k0+16..k0+23}, lanes 16-31 the other halves
        FragBF a;
        const unsigned short* ap = arow + k0 + half * 8;
        a.q[0] = *(const uint4*)(ap);
        a.q[1] = *(const uint4*)(ap + 16);
#pragma unroll
        for (int j = 0; j < NT; ++j) {
            // B fragment: col = mr, K {k0+half*8..+7, k0+16+half*8..+7} (matches A)
            const unsigned short* bp = W + (size_t)(n0 + j * 16 + mr) * K + k0 + half * 8;
            FragBF b;
            b.q[0] = *(const uint4*)(bp);
            b.q[1] = *(const uint4*)(bp + 16);
            __builtin_prefetch(bp + 32, 0, 0); // next K panel -> global_prefetch_b8
            acc[j].v = __builtin_amdgcn_wmma_f32_16x16x32_bf16(
                false, a.v, false, b.v, (short)0, acc[j].v, false, false);
        }
    }

    int rowoff = half * 8; // C layout: lanes 16-31 hold M rows 8..15
#pragma unroll
    for (int j = 0; j < NT; ++j) {
        int n = n0 + j * 16 + mr;
        float bv = bias ? bias[n] : 0.0f;
#pragma unroll
        for (int r = 0; r < 8; ++r) {
            int m = m0 + rowoff + r;
            C[(size_t)m * N + n] = acc[j].f[r] + bv;
        }
    }
}

// ---------- GRU cell (serial phase, fp32 VALU): one thread per (b, j) ----------
__global__ void k_gru(const float* __restrict__ emb, const int* __restrict__ target, int t,
                      const float* __restrict__ ch_prev, const float* __restrict__ h_prev,
                      const float* __restrict__ W_ih, const float* __restrict__ W_hh,
                      const float* __restrict__ b_ih, const float* __restrict__ b_hh,
                      float* __restrict__ h_new) {
    int tid = blockIdx.x * blockDim.x + threadIdx.x;
    if (tid >= Bn * Hn) return;
    int b = tid / Hn, j = tid % Hn;
    int tok = (t == 0) ? SOS : target[b * Tn + (t - 1)];
    const float* e   = emb + (size_t)tok * Hn;
    const float* chp = ch_prev + (size_t)b * Hn;
    const float* hp  = h_prev + (size_t)b * Hn;

    const float* wr = W_ih + (size_t)j * (2 * Hn);
    const float* wz = W_ih + (size_t)(Hn + j) * (2 * Hn);
    const float* wn = W_ih + (size_t)(2 * Hn + j) * (2 * Hn);
    float ir = b_ih[j], iz = b_ih[Hn + j], inn = b_ih[2 * Hn + j];
    for (int k = 0; k < Hn; ++k) { float x = e[k];   ir += x * wr[k];      iz += x * wz[k];      inn += x * wn[k]; }
    for (int k = 0; k < Hn; ++k) { float x = chp[k]; ir += x * wr[Hn + k]; iz += x * wz[Hn + k]; inn += x * wn[Hn + k]; }

    const float* vr = W_hh + (size_t)j * Hn;
    const float* vz = W_hh + (size_t)(Hn + j) * Hn;
    const float* vn = W_hh + (size_t)(2 * Hn + j) * Hn;
    float hr = b_hh[j], hz = b_hh[Hn + j], hn = b_hh[2 * Hn + j];
    for (int k = 0; k < Hn; ++k) { float hv = hp[k]; hr += hv * vr[k]; hz += hv * vz[k]; hn += hv * vn[k]; }

    float r = 1.0f / (1.0f + expf(-(ir + hr)));
    float z = 1.0f / (1.0f + expf(-(iz + hz)));
    float n = tanhf(inn + r * hn);
    h_new[tid] = (1.0f - z) * n + z * hp[j];
}

// ---------- attention scores + softmax, one block per batch row ----------
__global__ void k_attn(const float* __restrict__ proj, const float* __restrict__ h,
                       float* __restrict__ w_buf, float* attn_out, int t) {
    __shared__ float hs[Hn];
    __shared__ float red[Sn];
    int b = blockIdx.x, s = threadIdx.x; // blockDim == Sn == 256
    for (int k = s; k < Hn; k += Sn) hs[k] = h[b * Hn + k];
    __syncthreads();

    const float4* p4 = (const float4*)(proj + ((size_t)b * Sn + s) * Hn);
    float acc = 0.0f;
    for (int k = 0; k < Hn / 4; ++k) {
        float4 v = p4[k];
        acc += v.x * hs[4 * k] + v.y * hs[4 * k + 1] + v.z * hs[4 * k + 2] + v.w * hs[4 * k + 3];
    }
    red[s] = acc; __syncthreads();
    for (int o = Sn / 2; o > 0; o >>= 1) { if (s < o) red[s] = fmaxf(red[s], red[s + o]); __syncthreads(); }
    float mx = red[0]; __syncthreads();
    float ex = expf(acc - mx);
    red[s] = ex; __syncthreads();
    for (int o = Sn / 2; o > 0; o >>= 1) { if (s < o) red[s] += red[s + o]; __syncthreads(); }
    float wv = ex / red[0];
    w_buf[b * Sn + s] = wv;
    if (attn_out) attn_out[((size_t)b * Tn + t) * Sn + s] = wv;
}

// ---------- context: ctx[b,h] = sum_s w[b,s]*proj[b,s,h]; block per b ----------
__global__ void k_ctx(const float* __restrict__ proj, const float* __restrict__ w_buf,
                      float* __restrict__ ctx) {
    __shared__ float ws[Sn];
    int b = blockIdx.x, hc = threadIdx.x; // blockDim == Hn == 512
    for (int s = hc; s < Sn; s += blockDim.x) ws[s] = w_buf[b * Sn + s];
    __syncthreads();
    float acc = 0.0f;
    for (int s = 0; s < Sn; ++s) acc += ws[s] * proj[((size_t)b * Sn + s) * Hn + hc];
    ctx[b * Hn + hc] = acc;
}

// ---------- ch = tanh([ctx,h] @ Wc^T + bc); also record bf16 row for vocab GEMM ----------
__global__ void k_ch(const float* __restrict__ ctx, const float* __restrict__ h,
                     const float* __restrict__ Wc, const float* __restrict__ bc,
                     float* __restrict__ ch_f32, unsigned short* CH_bf16, int t) {
    int tid = blockIdx.x * blockDim.x + threadIdx.x;
    if (tid >= Bn * Hn) return;
    int b = tid / Hn, j = tid % Hn;
    const float* wc = Wc + (size_t)j * (2 * Hn);
    const float* cx = ctx + (size_t)b * Hn;
    const float* hb = h + (size_t)b * Hn;
    float acc = bc[j];
    for (int k = 0; k < Hn; ++k) acc += cx[k] * wc[k];
    for (int k = 0; k < Hn; ++k) acc += hb[k] * wc[Hn + k];
    float v = tanhf(acc);
    ch_f32[tid] = v;
    if (CH_bf16) CH_bf16[((size_t)b * Tn + t) * Hn + j] = f2bf(v);
}

// ---------- in-place log_softmax over V per row; block per row ----------
__global__ void k_logsoftmax(float* __restrict__ logits) {
    __shared__ float red[256];
    int row = blockIdx.x, tid = threadIdx.x;
    float* p = logits + (size_t)row * Vn;
    float mx = -INFINITY;
    for (int i = tid; i < Vn; i += 256) mx = fmaxf(mx, p[i]);
    red[tid] = mx; __syncthreads();
    for (int o = 128; o > 0; o >>= 1) { if (tid < o) red[tid] = fmaxf(red[tid], red[tid + o]); __syncthreads(); }
    mx = red[0]; __syncthreads();
    float sum = 0.0f;
    for (int i = tid; i < Vn; i += 256) sum += expf(p[i] - mx);
    red[tid] = sum; __syncthreads();
    for (int o = 128; o > 0; o >>= 1) { if (tid < o) red[tid] += red[tid + o]; __syncthreads(); }
    float ls = logf(red[0]) + mx; __syncthreads();
    for (int i = tid; i < Vn; i += 256) p[i] = p[i] - ls;
}

// ---------- host launch ----------
extern "C" void kernel_launch(void* const* d_in, const int* in_sizes, int n_in,
                              void* d_out, int out_size, void* d_ws, size_t ws_size,
                              hipStream_t stream) {
    const float* enc    = (const float*)d_in[0];   // (B,S,2H)
    const float* enc_h  = (const float*)d_in[1];   // (1,B,H)
    const int*   target = (const int*)d_in[2];     // (B,T)
    const float* emb    = (const float*)d_in[3];   // (V,H)
    const float* Wp     = (const float*)d_in[4];   // (H,2H)
    const float* bp     = (const float*)d_in[5];
    const float* W_ih   = (const float*)d_in[6];   // (3H,2H)
    const float* W_hh   = (const float*)d_in[7];   // (3H,H)
    const float* b_ih   = (const float*)d_in[8];
    const float* b_hh   = (const float*)d_in[9];
    const float* Wc     = (const float*)d_in[10];  // (H,2H)
    const float* bc     = (const float*)d_in[11];
    const float* Wo     = (const float*)d_in[12];  // (V,H)
    const float* bo     = (const float*)d_in[13];

    float* out = (float*)d_out;
    const size_t OUT_LOGP = 0;
    const size_t OUT_HT   = (size_t)Bn * Tn * Vn;
    const size_t OUT_ATTN = OUT_HT + (size_t)Bn * Hn;

    // workspace layout (256B aligned slabs)
    char* w = (char*)d_ws;
    size_t off = 0;
    auto alloc = [&](size_t bytes) { void* p = w + off; off = (off + bytes + 255) & ~(size_t)255; return p; };
    unsigned short* enc_bf = (unsigned short*)alloc((size_t)Bn * Sn * 2 * Hn * 2);
    unsigned short* Wp_bf  = (unsigned short*)alloc((size_t)Hn * 2 * Hn * 2);
    unsigned short* Wo_bf  = (unsigned short*)alloc((size_t)Vn * Hn * 2);
    float*          projf  = (float*)alloc((size_t)Bn * Sn * Hn * 4);
    unsigned short* CH_bf  = (unsigned short*)alloc((size_t)Bn * Tn * Hn * 2);
    float*          h_a    = (float*)alloc((size_t)Bn * Hn * 4);
    float*          h_b    = (float*)alloc((size_t)Bn * Hn * 4);
    float*          ch_f   = (float*)alloc((size_t)Bn * Hn * 4);
    float*          ctxf   = (float*)alloc((size_t)Bn * Hn * 4);
    float*          w_buf  = (float*)alloc((size_t)Bn * Sn * 4);
    (void)ws_size; (void)n_in; (void)in_sizes; (void)out_size;

    // Phase 0: bf16 conversions (graph-safe, deterministic)
    {
        int n1 = Bn * Sn * 2 * Hn;
        k_f32_to_bf16<<<(n1 + 255) / 256, 256, 0, stream>>>(enc, enc_bf, n1);
        int n2 = Hn * 2 * Hn;
        k_f32_to_bf16<<<(n2 + 255) / 256, 256, 0, stream>>>(Wp, Wp_bf, n2);
        int n3 = Vn * Hn;
        k_f32_to_bf16<<<(n3 + 255) / 256, 256, 0, stream>>>(Wo, Wo_bf, n3);
    }

    // Phase 1: proj = enc @ Wp^T + bp   (M=8192, N=512, K=1024) — WMMA bf16
    {
        int M = Bn * Sn, N = Hn, K = 2 * Hn;
        int waves = (M / 16) * (N / 64);
        k_wmma_gemm_bf16<<<waves / 8, 256, 0, stream>>>(enc_bf, Wp_bf, bp, projf, M, N, K);
    }

    // Phase 2: serial decode. h0 = encoder_hidden[0]; ch0 = attention(h0).
    k_copy_f32<<<(Bn * Hn + 255) / 256, 256, 0, stream>>>(enc_h, h_a, Bn * Hn);
    k_attn<<<Bn, Sn, 0, stream>>>(projf, h_a, w_buf, nullptr, 0);
    k_ctx<<<Bn, Hn, 0, stream>>>(projf, w_buf, ctxf);
    k_ch<<<(Bn * Hn + 255) / 256, 256, 0, stream>>>(ctxf, h_a, Wc, bc, ch_f, nullptr, 0);

    float* h_cur = h_a; float* h_nxt = h_b;
    for (int t = 0; t < Tn; ++t) {
        k_gru<<<(Bn * Hn + 255) / 256, 256, 0, stream>>>(emb, target, t, ch_f, h_cur,
                                                          W_ih, W_hh, b_ih, b_hh, h_nxt);
        k_attn<<<Bn, Sn, 0, stream>>>(projf, h_nxt, w_buf, out + OUT_ATTN, t);
        k_ctx<<<Bn, Hn, 0, stream>>>(projf, w_buf, ctxf);
        k_ch<<<(Bn * Hn + 255) / 256, 256, 0, stream>>>(ctxf, h_nxt, Wc, bc, ch_f, CH_bf, t);
        float* tmp = h_cur; h_cur = h_nxt; h_nxt = tmp;
    }
    k_copy_f32<<<(Bn * Hn + 255) / 256, 256, 0, stream>>>(h_cur, out + OUT_HT, Bn * Hn);

    // Phase 3: logits = CH @ Wo^T + bo  (M=2048, N=32000, K=512) — WMMA bf16, L2-resident Wo
    {
        int M = Bn * Tn, N = Vn, K = Hn;
        int waves = (M / 16) * (N / 64);
        k_wmma_gemm_bf16<<<waves / 8, 256, 0, stream>>>(CH_bf, Wo_bf, bo, out + OUT_LOGP, M, N, K);
    }
    // log_softmax in place over the (B*T, V) rows
    k_logsoftmax<<<Bn * Tn, 256, 0, stream>>>(out + OUT_LOGP);
}
